// FoveaModel_50027779064156
// MI455X (gfx1250) — compile-verified, matrix operand order checked by reading, not run on hardware
//
#include <hip/hip_runtime.h>
#include <hip/hip_bf16.h>

typedef __attribute__((ext_vector_type(16))) _Float16 v16h;
typedef __attribute__((ext_vector_type(8)))  float    v8f;

#define NPIX 224
#define IMG  (NPIX * NPIX)
#define BATCH 32

__device__ __forceinline__ float sigmoidf(float v) {
    return 1.0f / (1.0f + __expf(-v));
}

// ---------------------------------------------------------------------------
// Kernel 0: mask + fovea + hr.  fovea == r_t0, so write r_t0 into the fovea
// output slot; conv1 reads it back from d_out (no workspace needed).
// ---------------------------------------------------------------------------
__global__ __launch_bounds__(256) void mask_kernel(
    const float* __restrict__ x, const float* __restrict__ rand_mask,
    const int* __restrict__ samples,
    float* __restrict__ r_t0_fovea, float* __restrict__ hr_out) {
    int idx = blockIdx.x * 256 + threadIdx.x;
    if (idx >= BATCH * IMG) return;
    int b  = idx / IMG;
    int p  = idx - b * IMG;
    int y  = p / NPIX;
    int xx = p - y * NPIX;
    int s  = samples[b];          // samples shape (B,1), int32
    int px = s / 7, py = s % 7;
    bool patch = (y >= px * 32) & (y < px * 32 + 32) &
                 (xx >= py * 32) & (xx < py * 32 + 32);
    bool mask = (rand_mask[idx] > 0.95f) | patch;
#pragma unroll
    for (int c = 0; c < 3; ++c) {
        float xv = x[(b * 3 + c) * IMG + p];
        r_t0_fovea[(b * 3 + c) * IMG + p] = mask  ? xv : 0.0f;
        hr_out[(b * 3 + c) * IMG + p]     = patch ? xv : 0.0f;
    }
}

// ---------------------------------------------------------------------------
// Fused ConvLSTM step: 3x3 SAME conv over concat(inA[C1], inB[C2]) -> NOUT
// channels via v_wmma_f32_16x16x32_f16 (9 shifted 1x1 GEMMs, K padded to 32),
// then LSTM gates (f32) fused in the same block.
// One workgroup (8 waves) = one 16x16 spatial block of one image, all NOUT.
// ---------------------------------------------------------------------------
template <int C1, int C2, int NOUT, int CH>
__global__ __launch_bounds__(256) void convlstm_kernel(
    const float* __restrict__ inA, const float* __restrict__ inB,
    const float* __restrict__ cprev,
    const float* __restrict__ W, const float* __restrict__ Bias,
    float* __restrict__ h_out, float* __restrict__ c_out,
    float* __restrict__ h_out2) {
    constexpr int CIN = C1 + C2;
    constexpr int CP  = 32;                    // K padded to one WMMA K-step
    constexpr int NT  = (NOUT + 15) / 16;      // N tiles
    constexpr int NP  = NT * 16;
    constexpr int SIN_H = 18 * 18 * CP;                    // input tile halves
    constexpr int SW_H  = 9 * NP * CP;                     // weight halves
    constexpr int PH1_B = (SIN_H + SW_H) * 2;              // phase1/2 bytes
    constexpr int PH3_B = 256 * NP * 4;                    // z scratch bytes
    constexpr int SMEM_B = (PH1_B > PH3_B) ? PH1_B : PH3_B;

    __shared__ alignas(32) unsigned char smem[SMEM_B];
    _Float16* sIn = (_Float16*)smem;           // [ly][lx][c]  (18x18xCP)
    _Float16* sW  = sIn + SIN_H;               // [tap][n][c]
    float*    sZ  = (float*)smem;              // [pixel][n]  (aliased, phased)

    const int tid  = threadIdx.x;
    const int lane = tid & 31;
    const int wave = tid >> 5;
    const int lx0  = lane & 15;
    const int cbase = (lane >> 4) * 16;        // K half selector (0 / 16)

    const int bi  = blockIdx.x;                // B * 14 * 14 blocks
    const int b   = bi / 196;
    const int rem = bi - b * 196;
    const int by0 = (rem / 14) * 16;
    const int bx0 = (rem % 14) * 16;

    // ---- phase 1: stage input tile (with halo, zero padded) + weights ----
    for (int idx = tid; idx < SIN_H; idx += 256) {
        int c  = idx & (CP - 1);
        int p  = idx / CP;
        int lx = p % 18;
        int ly = p / 18;
        int gy = by0 + ly - 1;
        int gx = bx0 + lx - 1;
        float v = 0.0f;
        if (gy >= 0 && gy < NPIX && gx >= 0 && gx < NPIX && c < CIN) {
            v = (c < C1) ? inA[((b * C1 + c) * NPIX + gy) * NPIX + gx]
                         : inB[((b * C2 + (c - C1)) * NPIX + gy) * NPIX + gx];
        }
        sIn[idx] = (_Float16)v;
    }
    for (int idx = tid; idx < SW_H; idx += 256) {
        int c   = idx & (CP - 1);
        int n   = (idx / CP) % NP;
        int tap = idx / (CP * NP);
        float v = 0.0f;
        if (c < CIN && n < NOUT) v = W[(n * CIN + c) * 9 + tap];
        sW[idx] = (_Float16)v;
    }
    __syncthreads();

    // ---- phase 2: 9-tap implicit GEMM, all in registers ----
    v8f acc[2][NT] = {};
    for (int tap = 0; tap < 9; ++tap) {
        const int dy = tap / 3, dx = tap % 3;
        v16h bfr[NT];
#pragma unroll
        for (int nt = 0; nt < NT; ++nt)
            bfr[nt] = *(const v16h*)&sW[(tap * NP + nt * 16 + lx0) * CP + cbase];
#pragma unroll
        for (int mt = 0; mt < 2; ++mt) {
            const int ly = 2 * wave + mt + dy;
            v16h afr = *(const v16h*)&sIn[(ly * 18 + lx0 + dx) * CP + cbase];
#pragma unroll
            for (int nt = 0; nt < NT; ++nt)
                acc[mt][nt] = __builtin_amdgcn_wmma_f32_16x16x32_f16(
                    false, afr, false, bfr[nt], (short)0, acc[mt][nt],
                    false, false);
        }
    }
    __syncthreads();   // everyone done reading sIn/sW before sZ alias reuse

    // ---- phase 3: spill conv result z to LDS [pixel][n] ----
#pragma unroll
    for (int mt = 0; mt < 2; ++mt) {
        const int rowIdx = 2 * wave + mt;
#pragma unroll
        for (int nt = 0; nt < NT; ++nt) {
            const int n = nt * 16 + lx0;
#pragma unroll
            for (int r = 0; r < 8; ++r) {
                const int xpix = r + 8 * (lane >> 4);
                sZ[(rowIdx * 16 + xpix) * NP + n] = acc[mt][nt][r];
            }
        }
    }
    __syncthreads();

    // ---- phase 4: fused LSTM gates (f32), one pixel per thread ----
    const int pixel = tid;                 // 256 threads == 256 pixels
    const int gy = by0 + (pixel >> 4);
    const int gx = bx0 + (pixel & 15);
    for (int ch = 0; ch < CH; ++ch) {
        float iv = sigmoidf(sZ[pixel * NP + ch]          + Bias[ch]);
        float fv = sigmoidf(sZ[pixel * NP + CH + ch]     + Bias[CH + ch]);
        float ov = sigmoidf(sZ[pixel * NP + 2 * CH + ch] + Bias[2 * CH + ch]);
        float gv = tanhf(sZ[pixel * NP + 3 * CH + ch]    + Bias[3 * CH + ch]);
        const int gidx = ((b * CH + ch) * NPIX + gy) * NPIX + gx;
        float cn = fv * cprev[gidx] + iv * gv;
        float hn = ov * tanhf(cn);
        c_out[gidx] = cn;
        h_out[gidx] = hn;
        if (h_out2) h_out2[gidx] = hn;
    }
}

// ---------------------------------------------------------------------------
// Kernel 4: final 3->3 conv (reconstruction). 27 MACs/pixel, memory bound.
// ---------------------------------------------------------------------------
__global__ __launch_bounds__(256) void final_conv_kernel(
    const float* __restrict__ h, const float* __restrict__ wc,
    const float* __restrict__ bc, float* __restrict__ rec) {
    int idx = blockIdx.x * 256 + threadIdx.x;
    if (idx >= BATCH * 3 * IMG) return;
    int xx = idx % NPIX;
    int y  = (idx / NPIX) % NPIX;
    int co = (idx / IMG) % 3;
    int b  = idx / (3 * IMG);
    float acc = bc[co];
#pragma unroll
    for (int ci = 0; ci < 3; ++ci)
#pragma unroll
        for (int ky = 0; ky < 3; ++ky) {
            int gy = y + ky - 1;
            if (gy < 0 || gy >= NPIX) continue;
#pragma unroll
            for (int kx = 0; kx < 3; ++kx) {
                int gx = xx + kx - 1;
                if (gx < 0 || gx >= NPIX) continue;
                acc += wc[((co * 3 + ci) * 3 + ky) * 3 + kx] *
                       h[((b * 3 + ci) * NPIX + gy) * NPIX + gx];
            }
        }
    rec[idx] = acc;
}

// ---------------------------------------------------------------------------
extern "C" void kernel_launch(void* const* d_in, const int* in_sizes, int n_in,
                              void* d_out, int out_size, void* d_ws,
                              size_t ws_size, hipStream_t stream) {
    (void)in_sizes; (void)n_in; (void)out_size; (void)d_ws; (void)ws_size;
    const float* x         = (const float*)d_in[0];
    const float* h1        = (const float*)d_in[2];
    const float* c1        = (const float*)d_in[3];
    const float* h2        = (const float*)d_in[4];
    const float* c2        = (const float*)d_in[5];
    const float* h3        = (const float*)d_in[6];
    const float* c3        = (const float*)d_in[7];
    const float* rand_mask = (const float*)d_in[8];
    const int*   samples   = (const int*)d_in[9];
    const float* w1 = (const float*)d_in[10];
    const float* b1 = (const float*)d_in[11];
    const float* w2 = (const float*)d_in[12];
    const float* b2 = (const float*)d_in[13];
    const float* w3 = (const float*)d_in[14];
    const float* b3 = (const float*)d_in[15];
    const float* wc = (const float*)d_in[16];
    const float* bc = (const float*)d_in[17];

    float* out = (float*)d_out;
    const long long S1 = (long long)BATCH * 8 * IMG;
    const long long S2 = (long long)BATCH * 16 * IMG;
    const long long S3 = (long long)BATCH * 3 * IMG;
    float* o_h1n = out;
    float* o_c1n = out + S1;
    float* o_h2n = out + 2 * S1;
    float* o_c2n = out + 2 * S1 + S2;
    float* o_h3n = out + 2 * S1 + 2 * S2;
    float* o_c3n = o_h3n + S3;
    float* o_rec = o_h3n + 2 * S3;
    float* o_fov = o_h3n + 3 * S3;   // fovea == r_t0 (conv1 input)
    float* o_hr  = o_h3n + 4 * S3;
    float* o_hid = o_h3n + 5 * S3;   // hidden == h3n

    const int nblk = BATCH * 14 * 14;      // 16x16 spatial tiles per image

    mask_kernel<<<(BATCH * IMG + 255) / 256, 256, 0, stream>>>(
        x, rand_mask, samples, o_fov, o_hr);

    convlstm_kernel<3, 8, 32, 8><<<nblk, 256, 0, stream>>>(
        o_fov, h1, c1, w1, b1, o_h1n, o_c1n, nullptr);

    convlstm_kernel<8, 16, 64, 16><<<nblk, 256, 0, stream>>>(
        o_h1n, h2, c2, w2, b2, o_h2n, o_c2n, nullptr);

    convlstm_kernel<16, 3, 12, 3><<<nblk, 256, 0, stream>>>(
        o_h2n, h3, c3, w3, b3, o_h3n, o_c3n, o_hid);

    final_conv_kernel<<<(BATCH * 3 * IMG + 255) / 256, 256, 0, stream>>>(
        o_h3n, wc, bc, o_rec);
}
